// LSTM_54571854463188
// MI455X (gfx1250) — matrix operand order repaired and checked by reading, not compiled, not run on hardware
//
#include <hip/hip_runtime.h>
#include <stdint.h>

#define SEQ    512
#define BATCH  64
#define IN_DIM 512
#define HID    1024
#define G4     (4 * HID)   // 4096

// LDS row strides padded so fragment reads stripe across all 64 banks:
// stride(dwords) % 64 == 4  ->  lanes 0..15 cover banks 4m..4m+3 = all 64.
#define LDSH_STRIDE (HID + 8)      // 1032 bf16 = 516 dwords
#define LDSX_STRIDE (IN_DIM + 8)   // 520 bf16  = 260 dwords

typedef __bf16 bf16_t;
typedef __attribute__((ext_vector_type(16))) __bf16 v16bf;
typedef __attribute__((ext_vector_type(8)))  __bf16 v8bf;
typedef __attribute__((ext_vector_type(8)))  float  v8f;

// exact pointee type expected by the async-to-LDS builtin (from hipcc diag):
//   '__attribute__((__vector_size__(4 * sizeof(int)))) int *'
typedef int v4i_t __attribute__((__vector_size__(4 * sizeof(int))));
typedef v4i_t __attribute__((address_space(1))) as1_v4i;
typedef v4i_t __attribute__((address_space(3))) as3_v4i;

#if defined(__has_builtin)
#if __has_builtin(__builtin_amdgcn_global_load_async_to_lds_b128) && \
    __has_builtin(__builtin_amdgcn_s_wait_asynccnt)
#define USE_ASYNC_LDS 1
#endif
#endif
#ifndef USE_ASYNC_LDS
#define USE_ASYNC_LDS 0
#endif

union ABFrag { v16bf v; v8bf h[2]; };

// round-to-nearest-even f32 -> bf16
__device__ __forceinline__ bf16_t f2bf(float f) {
    union { float f; uint32_t u; } v; v.f = f;
    uint32_t r = v.u + 0x7FFFu + ((v.u >> 16) & 1u);
    uint16_t h = (uint16_t)(r >> 16);
    return __builtin_bit_cast(bf16_t, h);
}

__device__ __forceinline__ float sigmoidf_(float x) {
    return 1.0f / (1.0f + __expf(-x));
}

// --- WMMA fragment loaders (CDNA5 ISA 7.12.2 layouts, wave32) -------------
// A: 16x32 bf16 row-major, A pre-offset to tile row m0.
// lane l: row = l&15; chunk0 K = 8*(l>>4)..+7, chunk1 = +16
__device__ __forceinline__ v16bf load_a_frag(const bf16_t* A, int lda,
                                             int k0, int lane) {
    int m  = lane & 15;
    int kb = (lane >> 4) << 3;                 // 0 or 8
    const bf16_t* p = A + (size_t)m * lda + k0 + kb;
    ABFrag f;
    f.h[0] = *(const v8bf*)(p);
    f.h[1] = *(const v8bf*)(p + 16);
    return f.v;
}

// B: 32x16 bf16 where B(k,n) = W[n][k], W row-major [N,K]; W pre-offset to n0.
// lane l: col = l&15 (one W row); chunk0 K = 8*(l>>4)..+7, chunk1 = +16
__device__ __forceinline__ v16bf load_b_frag(const bf16_t* __restrict__ W,
                                             int ldw, int k0, int lane) {
    int n  = lane & 15;
    int kb = (lane >> 4) << 3;                 // 0 or 8
    const bf16_t* p = W + (size_t)n * ldw + k0 + kb;
    ABFrag f;
    f.h[0] = *(const v8bf*)(p);
    f.h[1] = *(const v8bf*)(p + 16);
    return f.v;
}

// 4 stacked M-tiles (rows m0, m0+16, m0+32, m0+48) x 1 N-tile:
// one B fragment feeds 4 WMMAs (4x cut in B traffic).
__device__ __forceinline__ void mm4(const bf16_t* A, int lda,
                                    const bf16_t* __restrict__ Wrow, int ldw,
                                    int K, int lane, v8f* acc) {
    #pragma unroll 2
    for (int k = 0; k < K; k += 32) {
        v16bf b  = load_b_frag(Wrow, ldw, k, lane);
        v16bf a0 = load_a_frag(A,                     lda, k, lane);
        v16bf a1 = load_a_frag(A + (size_t)16 * lda,  lda, k, lane);
        v16bf a2 = load_a_frag(A + (size_t)32 * lda,  lda, k, lane);
        v16bf a3 = load_a_frag(A + (size_t)48 * lda,  lda, k, lane);
        acc[0] = __builtin_amdgcn_wmma_f32_16x16x32_bf16(false, a0, false, b, (short)0, acc[0], false, false);
        acc[1] = __builtin_amdgcn_wmma_f32_16x16x32_bf16(false, a1, false, b, (short)0, acc[1], false, false);
        acc[2] = __builtin_amdgcn_wmma_f32_16x16x32_bf16(false, a2, false, b, (short)0, acc[2], false, false);
        acc[3] = __builtin_amdgcn_wmma_f32_16x16x32_bf16(false, a3, false, b, (short)0, acc[3], false, false);
    }
}

// Stage a rows x cols bf16 panel from global into LDS (16B chunks/thread).
// gfx1250 async-to-LDS path when available; plain load+ds_store fallback.
__device__ __forceinline__ void stage_to_lds(const bf16_t* g, bf16_t* lds,
                                             int rows, int cols, int lds_stride,
                                             int tid) {
    int chunks_per_row = cols >> 3;            // 16B chunks
    int total = rows * chunks_per_row;
    for (int c = tid; c < total; c += 256) {
        int r = c / chunks_per_row;
        int q = c - r * chunks_per_row;
        bf16_t* src = const_cast<bf16_t*>(g) + (size_t)r * cols + (q << 3);
        bf16_t* dst = lds + (size_t)r * lds_stride + (q << 3);
#if USE_ASYNC_LDS
        __builtin_amdgcn_global_load_async_to_lds_b128(
            (as1_v4i*)src, (as3_v4i*)dst, 0, 0);
#else
        *(v8bf*)dst = *(const v8bf*)src;
#endif
    }
}

// --- Phase 1: x_proj = X(32768x512) @ W_ih^T + b ---------------------------
// One wave = 64 rows (4 M-tiles) x 16 cols; B fragment reused 4x.
__global__ void __launch_bounds__(256) xproj_gemm_kernel(
        const bf16_t* __restrict__ Xb, const bf16_t* __restrict__ Wih,
        const float* __restrict__ bias, float* __restrict__ xproj) {
    int wave = (blockIdx.x * 256 + threadIdx.x) >> 5;
    int lane = threadIdx.x & 31;
    int tn = wave & (G4 / 16 - 1);             // 0..255
    int tg = wave >> 8;                        // 0..511 groups of 64 rows
    int m0 = tg * 64, n0 = tn * 16;

    v8f acc[4] = {};
    mm4(Xb + (size_t)m0 * IN_DIM, IN_DIM,
        Wih + (size_t)n0 * IN_DIM, IN_DIM, IN_DIM, lane, acc);

    int col = lane & 15;
    int rb  = (lane >> 4) << 3;
    float bv = bias[n0 + col];
    #pragma unroll
    for (int t4 = 0; t4 < 4; ++t4) {
        float* out = xproj + (size_t)(m0 + t4 * 16 + rb) * G4 + n0 + col;
        #pragma unroll
        for (int r = 0; r < 8; ++r) out[(size_t)r * G4] = acc[t4][r] + bv;
    }
}

// --- Per-step gates: gates = [x_t @ W_ih^T +] h @ W_hh^T + b [+ xproj_t] ---
// Block = 8 waves; h (and x_t when fused) staged once in LDS and shared.
// Wave w: cols n0 = blockIdx*128 + w*16, all 64 batch rows (4 accumulators).
__global__ void __launch_bounds__(256) step_gates_kernel(
        const bf16_t* __restrict__ h_bf, const bf16_t* __restrict__ Whh,
        const bf16_t* __restrict__ x_t,  const bf16_t* __restrict__ Wih,
        const float*  __restrict__ xproj_t, const float* __restrict__ bias,
        float* __restrict__ gates, int fused) {
    extern __shared__ bf16_t smem[];
    bf16_t* lds_h = smem;                            // BATCH x LDSH_STRIDE
    bf16_t* lds_x = smem + BATCH * LDSH_STRIDE;      // BATCH x LDSX_STRIDE

    int tid = threadIdx.x;
    stage_to_lds(h_bf, lds_h, BATCH, HID, LDSH_STRIDE, tid);
    if (fused) stage_to_lds(x_t, lds_x, BATCH, IN_DIM, LDSX_STRIDE, tid);
#if USE_ASYNC_LDS
    __builtin_amdgcn_s_wait_asynccnt(0);
#endif
    __syncthreads();

    int wave = tid >> 5, lane = tid & 31;
    int n0 = blockIdx.x * 128 + wave * 16;

    v8f acc[4] = {};
    if (fused)
        mm4(lds_x, LDSX_STRIDE, Wih + (size_t)n0 * IN_DIM, IN_DIM,
            IN_DIM, lane, acc);
    mm4(lds_h, LDSH_STRIDE, Whh + (size_t)n0 * HID, HID, HID, lane, acc);

    int col = lane & 15;
    int rb  = (lane >> 4) << 3;
    float bv = bias[n0 + col];
    #pragma unroll
    for (int t4 = 0; t4 < 4; ++t4) {
        size_t base = (size_t)(t4 * 16 + rb) * G4 + n0 + col;
        #pragma unroll
        for (int r = 0; r < 8; ++r) {
            float v = acc[t4][r] + bv;
            if (xproj_t) v += xproj_t[base + (size_t)r * G4];
            gates[base + (size_t)r * G4] = v;
        }
    }
}

// --- Per-step elementwise gating ------------------------------------------
__global__ void __launch_bounds__(256) step_elem_kernel(
        const float* __restrict__ gates, float* __restrict__ c,
        bf16_t* __restrict__ h_bf, float* __restrict__ h_out) {
    int idx = blockIdx.x * 256 + threadIdx.x;  // 65536 = B*H
    int b = idx >> 10, j = idx & (HID - 1);
    const float* gr = gates + (size_t)b * G4 + j;
    float gi = sigmoidf_(gr[0]);
    float gf = sigmoidf_(gr[HID]);
    float gg = tanhf(gr[2 * HID]);
    float go = sigmoidf_(gr[3 * HID]);
    float cn = gf * c[idx] + gi * gg;
    c[idx] = cn;
    float hn = go * tanhf(cn);
    h_out[idx] = hn;
    h_bf[idx]  = f2bf(hn);
}

__global__ void f32_to_bf16_kernel(const float* __restrict__ in,
                                   bf16_t* __restrict__ out, size_t n) {
    size_t i = (size_t)blockIdx.x * blockDim.x + threadIdx.x;
    size_t stride = (size_t)gridDim.x * blockDim.x;
    for (; i < n; i += stride) out[i] = f2bf(in[i]);
}

__global__ void init_state_kernel(const float* __restrict__ h0,
                                  const float* __restrict__ c0,
                                  bf16_t* __restrict__ h_bf,
                                  float* __restrict__ c) {
    int idx = blockIdx.x * 256 + threadIdx.x;
    h_bf[idx] = f2bf(h0[idx]);
    c[idx]    = c0[idx];
}

__global__ void finalize_kernel(const float* __restrict__ h_last,
                                const float* __restrict__ c,
                                float* __restrict__ hT, float* __restrict__ cT) {
    int idx = blockIdx.x * 256 + threadIdx.x;
    hT[idx] = h_last[idx];
    cT[idx] = c[idx];
}

extern "C" void kernel_launch(void* const* d_in, const int* in_sizes, int n_in,
                              void* d_out, int out_size, void* d_ws, size_t ws_size,
                              hipStream_t stream) {
    const float* x    = (const float*)d_in[0];
    const float* W_ih = (const float*)d_in[1];
    const float* W_hh = (const float*)d_in[2];
    const float* b    = (const float*)d_in[3];
    const float* h0   = (const float*)d_in[4];
    const float* c0   = (const float*)d_in[5];

    float* out   = (float*)d_out;
    float* h_seq = out;                                    // (SEQ,B,H)
    float* hT    = out + (size_t)SEQ * BATCH * HID;        // (1,B,H)
    float* cT    = hT + (size_t)BATCH * HID;               // (1,B,H)

    auto align = [](size_t o) { return (o + 255) & ~(size_t)255; };
    char* ws = (char*)d_ws;
    size_t off = 0;
    bf16_t* x_bf   = (bf16_t*)(ws + off); off = align(off + (size_t)SEQ * BATCH * IN_DIM * sizeof(bf16_t));
    bf16_t* Wih_bf = (bf16_t*)(ws + off); off = align(off + (size_t)G4 * IN_DIM * sizeof(bf16_t));
    bf16_t* Whh_bf = (bf16_t*)(ws + off); off = align(off + (size_t)G4 * HID * sizeof(bf16_t));
    float*  gates  = (float*)(ws + off);  off = align(off + (size_t)BATCH * G4 * sizeof(float));
    bf16_t* h_bf   = (bf16_t*)(ws + off); off = align(off + (size_t)BATCH * HID * sizeof(bf16_t));
    float*  c_st   = (float*)(ws + off);  off = align(off + (size_t)BATCH * HID * sizeof(float));
    float*  xproj  = (float*)(ws + off);
    size_t xproj_bytes = (size_t)SEQ * BATCH * G4 * sizeof(float);  // 512 MB
    bool pathA = (ws_size >= off + xproj_bytes);           // precompute x_proj?

    // one-time conversions / state init
    f32_to_bf16_kernel<<<2048, 256, 0, stream>>>(x,    x_bf,   (size_t)SEQ * BATCH * IN_DIM);
    f32_to_bf16_kernel<<<2048, 256, 0, stream>>>(W_ih, Wih_bf, (size_t)G4 * IN_DIM);
    f32_to_bf16_kernel<<<2048, 256, 0, stream>>>(W_hh, Whh_bf, (size_t)G4 * HID);
    init_state_kernel<<<(BATCH * HID) / 256, 256, 0, stream>>>(h0, c0, h_bf, c_st);

    if (pathA) {
        int waves = (SEQ * BATCH / 64) * (G4 / 16);        // 131072 (4 M-tiles/wave)
        xproj_gemm_kernel<<<waves / 8, 256, 0, stream>>>(x_bf, Wih_bf, b, xproj);
    }

    const int fused = pathA ? 0 : 1;
    const size_t shmem = (size_t)BATCH * LDSH_STRIDE * sizeof(bf16_t) +
                         (fused ? (size_t)BATCH * LDSX_STRIDE * sizeof(bf16_t) : 0);
    const int gblocks = G4 / 128;                          // 32 blocks x 8 waves
    for (int t = 0; t < SEQ; ++t) {
        const bf16_t* x_t  = pathA ? nullptr : x_bf + (size_t)t * BATCH * IN_DIM;
        const float*  xp_t = pathA ? xproj + (size_t)t * BATCH * G4 : nullptr;
        step_gates_kernel<<<gblocks, 256, shmem, stream>>>(
            h_bf, Whh_bf, x_t, Wih_bf, xp_t, b, gates, fused);
        step_elem_kernel<<<(BATCH * HID) / 256, 256, 0, stream>>>(
            gates, c_st, h_bf, h_seq + (size_t)t * BATCH * HID);
    }
    finalize_kernel<<<(BATCH * HID) / 256, 256, 0, stream>>>(
        h_seq + (size_t)(SEQ - 1) * BATCH * HID, c_st, hT, cT);
}
